// TrendAwareAttention_8495445311638
// MI455X (gfx1250) — matrix operand order repaired and checked by reading, not compiled, not run on hardware
//
#include <hip/hip_runtime.h>
#include <hip/hip_bf16.h>

// ---------------------------------------------------------------------------
// TrendAwareAttention for MI455X (gfx1250), wave32 + v_wmma_f32_16x16x32_f16
//
// Pipeline (all f16 GEMMs, f32 accumulate, staged through the 192MB L2):
//   prep   : weights -> f16, conv weights rearranged to [tap][o][c]
//   proj<3>: Q = conv1d(query,Wq)+bq   -> ws f16 [bn][t][d]
//   proj<3>: K = conv1d(key,Wk)+bk     -> ws f16 [bn][t][d]
//   proj<1>: V = value@Wv^T+bv         -> ws f16 [bn][d][t]  (transposed)
//   attn   : per (b,n,h): S=QK^T/sqrt(dk), mask, softmax, P@V -> ws f16
//   outproj: out = O@Wo^T + bo         -> d_out f32
// ---------------------------------------------------------------------------

typedef _Float16 f16;
typedef __attribute__((ext_vector_type(16))) _Float16 v16h;
typedef __attribute__((ext_vector_type(8)))  _Float16 v8h;
typedef __attribute__((ext_vector_type(8)))  float    v8f;

#define BN_TOTAL 256   // B*N
#define TT       128   // TQ == TK
#define DM       256   // d_model
#define NH       8
#define DKH      32    // d_model / heads

__device__ inline v8f zero8() {
    v8f z;
#pragma unroll
    for (int i = 0; i < 8; ++i) z[i] = 0.0f;
    return z;
}

// A-fragment (16x32 f16) from an f32 row-major matrix.
// contiguous k order used here (B matches): lane m = lane%16; element e -> k = (lane/16)*16 + e.
__device__ inline v16h a_frag_from_f32(const float* rowptr, int kbase, int half) {
    v16h a;
    if (rowptr) {
        v8f lo = *(const v8f*)(rowptr + kbase + half * 16);
        v8f hi = *(const v8f*)(rowptr + kbase + half * 16 + 8);
#pragma unroll
        for (int e = 0; e < 8; ++e) { a[e] = (f16)lo[e]; a[e + 8] = (f16)hi[e]; }
    } else {
#pragma unroll
        for (int e = 0; e < 16; ++e) a[e] = (f16)0.0f;
    }
    return a;
}

// A-fragment from an f16 row-major matrix (global or LDS).
__device__ inline v16h a_frag_from_f16(const f16* rowptr, int kbase, int half) {
    v8h lo = *(const v8h*)(rowptr + kbase + half * 16);
    v8h hi = *(const v8h*)(rowptr + kbase + half * 16 + 8);
    v16h a;
#pragma unroll
    for (int e = 0; e < 8; ++e) { a[e] = lo[e]; a[e + 8] = hi[e]; }
    return a;
}

// ---------------------------------------------------------------------------
// Weight prep: f32 -> f16, conv weights (o,c,tap) -> [tap][o][c].
// ---------------------------------------------------------------------------
__global__ void taa_prep_weights(const float* __restrict__ Wq, const float* __restrict__ Wk,
                                 const float* __restrict__ Wv, const float* __restrict__ Wo,
                                 f16* __restrict__ wq16, f16* __restrict__ wk16,
                                 f16* __restrict__ wv16, f16* __restrict__ wo16) {
    int i = blockIdx.x * blockDim.x + threadIdx.x;
    if (i < 3 * DM * DM) {
        int tap = i / (DM * DM);
        int rem = i - tap * (DM * DM);
        int o = rem >> 8, c = rem & 255;
        size_t src = (size_t)(o * DM + c) * 3 + tap;
        wq16[i] = (f16)Wq[src];
        wk16[i] = (f16)Wk[src];
    }
    if (i < DM * DM) {
        wv16[i] = (f16)Wv[i];
        wo16[i] = (f16)Wo[i];
    }
}

// ---------------------------------------------------------------------------
// Projection (conv1d for TAPS=3, plain GEMM for TAPS=1).
// grid.x = BN*2 (64-row slabs), 256 threads (8 waves).
// Wave w: t-tile = slab*4 + w/2, o-tiles = (w&1)*8 .. +7.
// Y layout: transposed==0 -> [bn][t][o] f16 ; transposed==1 -> [bn][o][t] f16.
// ---------------------------------------------------------------------------
template <int TAPS>
__global__ void taa_proj(const float* __restrict__ X, const f16* __restrict__ W16,
                         const float* __restrict__ bias, f16* __restrict__ Y,
                         int transposed) {
    int wg   = blockIdx.x;
    int bn   = wg >> 1;
    int slab = wg & 1;
    int wave = threadIdx.x >> 5;
    int lane = threadIdx.x & 31;
    int half = lane >> 4, n16 = lane & 15;
    int t0     = (slab * 4 + (wave >> 1)) * 16;
    int otbase = (wave & 1) * 8;

    const float* Xseq = X + (size_t)bn * TT * DM;

    v8f acc[8];
#pragma unroll
    for (int i = 0; i < 8; ++i) acc[i] = zero8();

#pragma unroll
    for (int tap = 0; tap < TAPS; ++tap) {
        int dt  = (TAPS == 3) ? (tap - 1) : 0;
        int row = t0 + n16 + dt;
        const float* rowptr = (row >= 0 && row < TT) ? (Xseq + (size_t)row * DM) : nullptr;
        const f16* Wt = W16 + (size_t)tap * DM * DM;
#pragma unroll
        for (int kc = 0; kc < 8; ++kc) {            // 8 chunks of 32 input channels
            int cbase = kc * 32;
            v16h a = a_frag_from_f32(rowptr, cbase, half);
#pragma unroll
            for (int ot = 0; ot < 8; ++ot) {
                int o0 = (otbase + ot) * 16;
                // B-fragment: b[e] = W[o0+lane%16][cbase + (lane/16)*16 + e]
                v16h b = *(const v16h*)(Wt + (size_t)(o0 + n16) * DM + cbase + half * 16);
                acc[ot] = __builtin_amdgcn_wmma_f32_16x16x32_f16(
                    false, a, false, b, (short)0, acc[ot], false, false);
            }
        }
    }

#pragma unroll
    for (int ot = 0; ot < 8; ++ot) {
        int o0 = (otbase + ot) * 16;
        float bv = bias[o0 + n16];
        if (!transposed) {
            // D layout: lane holds rows m = r + 8*half, col n16 -> column stores
#pragma unroll
            for (int r = 0; r < 8; ++r) {
                int trow = t0 + r + 8 * half;
                Y[(size_t)bn * TT * DM + (size_t)trow * DM + o0 + n16] = (f16)(acc[ot][r] + bv);
            }
        } else {
            // transposed: row o = o0+n16, 8 consecutive t -> one 16B store
            v8h h;
#pragma unroll
            for (int r = 0; r < 8; ++r) h[r] = (f16)(acc[ot][r] + bv);
            *(v8h*)(Y + (size_t)bn * DM * TT + (size_t)(o0 + n16) * TT + t0 + 8 * half) = h;
        }
    }
}

// ---------------------------------------------------------------------------
// Attention: grid.x = BN*NH, 256 threads (8 waves). Wave w owns q-rows [16w,16w+16).
// ---------------------------------------------------------------------------
__global__ void taa_attn(const f16* __restrict__ Q, const f16* __restrict__ K,
                         const f16* __restrict__ Vt, const unsigned char* __restrict__ mask,
                         f16* __restrict__ O) {
    __shared__ f16 P[8][16][136];   // per-wave 16x128 prob tile (padded pitch)

    int blk  = blockIdx.x;
    int bn   = blk >> 3;
    int h    = blk & 7;
    int wave = threadIdx.x >> 5;
    int lane = threadIdx.x & 31;
    int half = lane >> 4, n16 = lane & 15;
    int t0 = wave * 16;

    const f16* Qs = Q  + (size_t)bn * TT * DM + h * DKH;
    const f16* Ks = K  + (size_t)bn * TT * DM + h * DKH;
    const f16* Vs = Vt + (size_t)bn * DM * TT + (size_t)h * DKH * TT;
    const unsigned char* mrow = mask + (size_t)bn * TT;

    // Q fragment: A[m = t0+lane%16][k = head dim 0..31]
    v16h aq = a_frag_from_f16(Qs + (size_t)(t0 + n16) * DM, 0, half);

    // scores: 8 tiles along tk, one 16x16x32 WMMA each
    v8f sacc[8];
#pragma unroll
    for (int T = 0; T < 8; ++T) {
        v16h bk = *(const v16h*)(Ks + (size_t)(T * 16 + n16) * DM + half * 16);
        sacc[T] = __builtin_amdgcn_wmma_f32_16x16x32_f16(
            false, aq, false, bk, (short)0, zero8(), false, false);
    }

    // key-padding mask for this lane's columns tk = T*16 + n16
    bool mbit[8];
#pragma unroll
    for (int T = 0; T < 8; ++T) mbit[T] = (mrow[T * 16 + n16] != 0);

    const float scale = 0.1767766952966369f;   // 1/sqrt(32)

    // row-wise softmax; row m = r + 8*half lives in one 16-lane half-wave
#pragma unroll
    for (int r = 0; r < 8; ++r) {
        float v[8];
        float mx = -3.0e38f;
#pragma unroll
        for (int T = 0; T < 8; ++T) {
            float s = sacc[T][r] * scale;
            if (mbit[T]) s = -1.0e9f;
            v[T] = s;
            mx = fmaxf(mx, s);
        }
#pragma unroll
        for (int off = 1; off < 16; off <<= 1) mx = fmaxf(mx, __shfl_xor(mx, off, 32));
        float sum = 0.0f;
#pragma unroll
        for (int T = 0; T < 8; ++T) { float e = __expf(v[T] - mx); v[T] = e; sum += e; }
#pragma unroll
        for (int off = 1; off < 16; off <<= 1) sum += __shfl_xor(sum, off, 32);
        float inv = 1.0f / sum;
#pragma unroll
        for (int T = 0; T < 8; ++T)
            P[wave][r + 8 * half][T * 16 + n16] = (f16)(v[T] * inv);
    }
    // wave-private LDS block: same-wave DS ordering handled by compiler waits

    // P @ V_h : out 16x32 (2 tiles), k over Tk=128 in 4 chunks of 32
    v8f oacc[2];
    oacc[0] = zero8(); oacc[1] = zero8();
#pragma unroll
    for (int kc = 0; kc < 4; ++kc) {
        v16h ap = a_frag_from_f16(&P[wave][n16][0], kc * 32, half);
#pragma unroll
        for (int dc = 0; dc < 2; ++dc) {
            // b[e] = V[tk = kc*32 + (lane/16)*16 + e][d = dc*16 + lane%16], via Vt[d][t]
            v16h bv = *(const v16h*)(Vs + (size_t)(dc * 16 + n16) * TT + kc * 32 + half * 16);
            oacc[dc] = __builtin_amdgcn_wmma_f32_16x16x32_f16(
                false, ap, false, bv, (short)0, oacc[dc], false, false);
        }
    }

    // store head output: O[bn][t][h*32 + d]
    f16* Os = O + (size_t)bn * TT * DM + h * DKH;
#pragma unroll
    for (int dc = 0; dc < 2; ++dc)
#pragma unroll
        for (int r = 0; r < 8; ++r)
            Os[(size_t)(t0 + r + 8 * half) * DM + dc * 16 + n16] = (f16)oacc[dc][r];
}

// ---------------------------------------------------------------------------
// Output projection: out = O @ Wo^T + bo (f32 result).
// ---------------------------------------------------------------------------
__global__ void taa_outproj(const f16* __restrict__ Oin, const f16* __restrict__ Wo16,
                            const float* __restrict__ bo, float* __restrict__ out) {
    int wg   = blockIdx.x;
    int bn   = wg >> 1;
    int slab = wg & 1;
    int wave = threadIdx.x >> 5;
    int lane = threadIdx.x & 31;
    int half = lane >> 4, n16 = lane & 15;
    int t0     = (slab * 4 + (wave >> 1)) * 16;
    int otbase = (wave & 1) * 8;

    const f16* Arow = Oin + (size_t)bn * TT * DM + (size_t)(t0 + n16) * DM;

    v8f acc[8];
#pragma unroll
    for (int i = 0; i < 8; ++i) acc[i] = zero8();

#pragma unroll
    for (int kc = 0; kc < 8; ++kc) {
        v16h a = a_frag_from_f16(Arow, kc * 32, half);
#pragma unroll
        for (int ot = 0; ot < 8; ++ot) {
            int o0 = (otbase + ot) * 16;
            v16h b = *(const v16h*)(Wo16 + (size_t)(o0 + n16) * DM + kc * 32 + half * 16);
            acc[ot] = __builtin_amdgcn_wmma_f32_16x16x32_f16(
                false, a, false, b, (short)0, acc[ot], false, false);
        }
    }

#pragma unroll
    for (int ot = 0; ot < 8; ++ot) {
        int o0 = (otbase + ot) * 16;
        float bb = bo[o0 + n16];
#pragma unroll
        for (int r = 0; r < 8; ++r)
            out[(size_t)bn * TT * DM + (size_t)(t0 + r + 8 * half) * DM + o0 + n16] =
                acc[ot][r] + bb;
    }
}

// ---------------------------------------------------------------------------
extern "C" void kernel_launch(void* const* d_in, const int* in_sizes, int n_in,
                              void* d_out, int out_size, void* d_ws, size_t ws_size,
                              hipStream_t stream) {
    const float* query = (const float*)d_in[0];
    const float* key   = (const float*)d_in[1];
    const float* value = (const float*)d_in[2];
    const float* Wq    = (const float*)d_in[3];
    const float* bq    = (const float*)d_in[4];
    const float* Wk    = (const float*)d_in[5];
    const float* bk    = (const float*)d_in[6];
    const float* Wv    = (const float*)d_in[7];
    const float* bv    = (const float*)d_in[8];
    const float* Wo    = (const float*)d_in[9];
    const float* bo    = (const float*)d_in[10];
    const unsigned char* mask = (const unsigned char*)d_in[11];
    float* out = (float*)d_out;

    char* ws = (char*)d_ws;
    // workspace layout (bytes):
    f16* wq16 = (f16*)(ws + 0);                         // 3*256*256*2 = 393216
    f16* wk16 = (f16*)(ws + 393216);                    // 393216
    f16* wv16 = (f16*)(ws + 786432);                    // 131072
    f16* wo16 = (f16*)(ws + 917504);                    // 131072
    f16* Qws  = (f16*)(ws + 1048576);                   // 256*128*256*2 = 16777216
    f16* Kws  = (f16*)(ws + 1048576 + 16777216ull);
    f16* Vtws = (f16*)(ws + 1048576 + 2 * 16777216ull);
    f16* Ows  = (f16*)(ws + 1048576 + 3 * 16777216ull); // total ~68 MB

    taa_prep_weights<<<768, 256, 0, stream>>>(Wq, Wk, Wv, Wo, wq16, wk16, wv16, wo16);

    taa_proj<3><<<BN_TOTAL * 2, 256, 0, stream>>>(query, wq16, bq, Qws, 0);
    taa_proj<3><<<BN_TOTAL * 2, 256, 0, stream>>>(key,   wk16, bk, Kws, 0);
    taa_proj<1><<<BN_TOTAL * 2, 256, 0, stream>>>(value, wv16, bv, Vtws, 1);

    taa_attn<<<BN_TOTAL * NH, 256, 0, stream>>>(Qws, Kws, Vtws, mask, Ows);

    taa_outproj<<<BN_TOTAL * 2, 256, 0, stream>>>(Ows, wo16, bo, out);
}
